// SimpleWaveGNN_12532714570222
// MI455X (gfx1250) — matrix operand order repaired and checked by reading, not compiled, not run on hardware
//
#include <hip/hip_runtime.h>
#include <hip/hip_bf16.h>

// ---------------------------------------------------------------------------
// SimpleWaveGNN on gfx1250 (MI455X): fused WMMA pipeline + async LDS gathers.
//   h = enc(x); L x { edge-MLP (2 fused GEMMs) -> atomic scatter-add agg;
//                    node-MLP (2 fused GEMMs) residual update }; decode.
// GEMMs: v_wmma_f32_16x16x32_f16. Edge/node/decoder gathers of h use a f16
// mirror (h16) moved directly global->LDS with GLOBAL_LOAD_ASYNC_TO_LDS_B128
// (ASYNCcnt / s_wait_asynccnt) -- no VGPR staging, half the L2 bytes.
// Workspace (~65 MB): h[N*64]f32 | agg[N*64]f32 | h16[N*64]f16 | f16 weights.
// ---------------------------------------------------------------------------

typedef __attribute__((ext_vector_type(16))) _Float16 v16h;
typedef __attribute__((ext_vector_type(8)))  float    v8f;

union HFrag { unsigned u[8]; v16h h; };

__device__ __forceinline__ v8f wmma_acc(v16h a, v16h b, v8f c) {
  return __builtin_amdgcn_wmma_f32_16x16x32_f16(false, a, false, b, (short)0, c,
                                                false, false);
}

// A-matrix fragment (16x32 f16, M x K) from a row-major [16 x K] LDS tile.
__device__ __forceinline__ v16h load_a_frag(const _Float16* tile, int K,
                                            int row, int k0, int hg) {
  HFrag f;
#pragma unroll
  for (int v = 0; v < 8; ++v) {
    int k = k0 + 2 * (v & 3) + ((v >> 2) << 4) + (hg << 3);
    f.u[v] = *(const unsigned*)(tile + row * K + k);
  }
  return f.h;
}

// B-matrix fragment (32x16 f16, K x N) from transposed weights Wt[out][K].
__device__ __forceinline__ v16h load_b_frag(const _Float16* __restrict__ wt,
                                            int K, int n0, int ln, int k0,
                                            int hg) {
  HFrag f;
#pragma unroll
  for (int v = 0; v < 8; ++v) {
    int k = k0 + 2 * (v & 3) + ((v >> 2) << 4) + (hg << 3);
    f.u[v] = *(const unsigned*)(wt + (size_t)(n0 + ln) * K + k);
  }
  return f.h;
}

__device__ __forceinline__ v8f vzero8() {
  v8f z = {0.f, 0.f, 0.f, 0.f, 0.f, 0.f, 0.f, 0.f};
  return z;
}

// Async global->LDS copy of 16 bytes (per-lane addresses). LDS offset is the
// low 32 bits of the generic address (flat aperture maps LDS at addr[31:0]).
__device__ __forceinline__ void async_g2l_b128(void* lds, const void* g) {
  unsigned l = (unsigned)(uintptr_t)lds;
  unsigned long long a = (unsigned long long)(uintptr_t)g;
  asm volatile("global_load_async_to_lds_b128 %0, %1, off" ::"v"(l), "v"(a)
               : "memory");
}

#define LDS_FENCE()   asm volatile("s_wait_dscnt 0" ::: "memory")
#define GATHER_FENCE() \
  asm volatile("s_wait_asynccnt 0\n\ts_wait_dscnt 0" ::: "memory")

// ---------------------------------------------------------------------------
// Weight convert+transpose: w[in][out] f32 -> wt[out][in_pad] f16 (zero pad).
// ---------------------------------------------------------------------------
__global__ void conv_wt_kernel(const float* __restrict__ w,
                               _Float16* __restrict__ wt, int in_dim,
                               int out_dim, int in_pad) {
  int i = blockIdx.x * blockDim.x + threadIdx.x;
  if (i >= out_dim * in_pad) return;
  int o = i / in_pad, k = i % in_pad;
  wt[i] = (k < in_dim) ? (_Float16)w[(size_t)k * out_dim + o] : (_Float16)0.f;
}

__global__ void zero_f32_kernel(float* __restrict__ p, long long n) {
  long long i = (long long)blockIdx.x * blockDim.x + threadIdx.x;
  if (i < n) p[i] = 0.f;
}

// ---------------------------------------------------------------------------
// Encoder: h = relu(x @ W1 + b1) @ W2 + b2.   x:[N,8] padded K->32.
// Writes f32 h (residual stream) and f16 mirror h16 (GEMM operand stream).
// ---------------------------------------------------------------------------
__global__ void __launch_bounds__(256) encode_kernel(
    const float* __restrict__ x, const _Float16* __restrict__ w1t,  // [64][32]
    const float* __restrict__ b1, const _Float16* __restrict__ w2t, // [64][64]
    const float* __restrict__ b2, float* __restrict__ h,
    _Float16* __restrict__ h16, int n_nodes) {
  __shared__ _Float16 Atile[8][16 * 32];
  __shared__ _Float16 Mtile[8][16 * 64];
  int wave = threadIdx.x >> 5, lane = threadIdx.x & 31;
  int ln = lane & 15, hg = lane >> 4;
  int tile = blockIdx.x * 8 + wave, base = tile * 16;
  if (base >= n_nodes) return;
  _Float16* A = Atile[wave];
  _Float16* M = Mtile[wave];

  int node = base + ln;
#pragma unroll
  for (int c = 0; c < 16; ++c) {
    int gc = hg * 16 + c;
    float v = (gc < 8) ? x[(size_t)node * 8 + gc] : 0.f;
    A[ln * 32 + gc] = (_Float16)v;
  }
  LDS_FENCE();

  v8f acc[4];
#pragma unroll
  for (int n = 0; n < 4; ++n) acc[n] = vzero8();
  {
    v16h af = load_a_frag(A, 32, ln, 0, hg);
#pragma unroll
    for (int n = 0; n < 4; ++n)
      acc[n] = wmma_acc(af, load_b_frag(w1t, 32, n * 16, ln, 0, hg), acc[n]);
  }
#pragma unroll
  for (int n = 0; n < 4; ++n) {
    float bias = b1[n * 16 + ln];
#pragma unroll
    for (int v = 0; v < 8; ++v) {
      float t = acc[n][v] + bias;
      M[(v + 8 * hg) * 64 + n * 16 + ln] = (_Float16)fmaxf(t, 0.f);
    }
  }
  LDS_FENCE();

  v8f acc2[4];
#pragma unroll
  for (int n = 0; n < 4; ++n) acc2[n] = vzero8();
#pragma unroll
  for (int k0 = 0; k0 < 64; k0 += 32) {
    v16h af = load_a_frag(M, 64, ln, k0, hg);
#pragma unroll
    for (int n = 0; n < 4; ++n)
      acc2[n] = wmma_acc(af, load_b_frag(w2t, 64, n * 16, ln, k0, hg), acc2[n]);
  }
#pragma unroll
  for (int n = 0; n < 4; ++n) {
    float bias = b2[n * 16 + ln];
#pragma unroll
    for (int v = 0; v < 8; ++v) {
      int row = v + 8 * hg;
      size_t idx = (size_t)(base + row) * 64 + n * 16 + ln;
      float t = acc2[n][v] + bias;
      h[idx] = t;
      h16[idx] = (_Float16)t;
    }
  }
}

// ---------------------------------------------------------------------------
// Edge message kernel: msg = relu([h[dst]|h[src]|ea] @ W1 + b1) @ W2 + b2,
// scatter-add into agg[dst]. 16 edges/wave; h rows DMA'd async from h16.
// ---------------------------------------------------------------------------
#define EK 160
__global__ void __launch_bounds__(256) edge_msg_kernel(
    const _Float16* __restrict__ h16, const int* __restrict__ src,
    const int* __restrict__ dst, const float* __restrict__ eattr,
    const _Float16* __restrict__ w1t,  // [64][160]
    const float* __restrict__ b1,
    const _Float16* __restrict__ w2t,  // [64][64]
    const float* __restrict__ b2, float* __restrict__ agg, int n_edges) {
  __shared__ _Float16 Atile[8][16 * EK];
  __shared__ _Float16 Mtile[8][16 * 64];
  int wave = threadIdx.x >> 5, lane = threadIdx.x & 31;
  int ln = lane & 15, hg = lane >> 4;
  int tile = blockIdx.x * 8 + wave, e0 = tile * 16;
  if (e0 >= n_edges) return;
  _Float16* A = Atile[wave];
  _Float16* M = Mtile[wave];

  __builtin_prefetch(w1t, 0, 0);  // weights are the hot L0/L2 operand

  // Gather: hg==0 -> x_i = h[dst] (cols 0..63); hg==1 -> x_j = h[src]
  // (cols 64..127) plus edge_attr (128..130) and zero pad (131..159).
  {
    int e = e0 + ln;
    int node = hg ? src[e] : dst[e];
    const _Float16* gp = h16 + (size_t)node * 64;
    _Float16* lp = A + ln * EK + hg * 64;
#pragma unroll
    for (int c = 0; c < 64; c += 8) async_g2l_b128(lp + c, gp + c);
    if (hg) {
      A[ln * EK + 128] = (_Float16)eattr[(size_t)e * 3 + 0];
      A[ln * EK + 129] = (_Float16)eattr[(size_t)e * 3 + 1];
      A[ln * EK + 130] = (_Float16)eattr[(size_t)e * 3 + 2];
#pragma unroll
      for (int c = 131; c < EK; ++c) A[ln * EK + c] = (_Float16)0.f;
    }
  }
  GATHER_FENCE();

  // GEMM1: [16 x 160] @ [160 x 64]
  v8f acc[4];
#pragma unroll
  for (int n = 0; n < 4; ++n) acc[n] = vzero8();
#pragma unroll
  for (int k0 = 0; k0 < EK; k0 += 32) {
    v16h af = load_a_frag(A, EK, ln, k0, hg);
#pragma unroll
    for (int n = 0; n < 4; ++n)
      acc[n] = wmma_acc(af, load_b_frag(w1t, EK, n * 16, ln, k0, hg), acc[n]);
  }
#pragma unroll
  for (int n = 0; n < 4; ++n) {
    float bias = b1[n * 16 + ln];
#pragma unroll
    for (int v = 0; v < 8; ++v) {
      float t = acc[n][v] + bias;
      M[(v + 8 * hg) * 64 + n * 16 + ln] = (_Float16)fmaxf(t, 0.f);
    }
  }
  LDS_FENCE();

  // GEMM2: [16 x 64] @ [64 x 64], then scatter-add to agg[dst].
  v8f acc2[4];
#pragma unroll
  for (int n = 0; n < 4; ++n) acc2[n] = vzero8();
#pragma unroll
  for (int k0 = 0; k0 < 64; k0 += 32) {
    v16h af = load_a_frag(M, 64, ln, k0, hg);
#pragma unroll
    for (int n = 0; n < 4; ++n)
      acc2[n] = wmma_acc(af, load_b_frag(w2t, 64, n * 16, ln, k0, hg), acc2[n]);
  }
#pragma unroll
  for (int n = 0; n < 4; ++n) {
    float bias = b2[n * 16 + ln];
#pragma unroll
    for (int v = 0; v < 8; ++v) {
      int row = v + 8 * hg;
      int d = dst[e0 + row];
      atomicAdd(&agg[(size_t)d * 64 + n * 16 + ln], acc2[n][v] + bias);
    }
  }
}

// ---------------------------------------------------------------------------
// Node update: h += relu([h|agg] @ V1 + c1) @ V2 + c2.  K = 128.
// h half of the gather is async-DMA'd from h16; agg half is f32->f16.
// ---------------------------------------------------------------------------
__global__ void __launch_bounds__(256) node_upd_kernel(
    float* __restrict__ h, _Float16* __restrict__ h16,
    const float* __restrict__ agg,
    const _Float16* __restrict__ v1t,  // [64][128]
    const float* __restrict__ c1,
    const _Float16* __restrict__ v2t,  // [64][64]
    const float* __restrict__ c2, int n_nodes) {
  __shared__ _Float16 Atile[8][16 * 128];
  __shared__ _Float16 Mtile[8][16 * 64];
  int wave = threadIdx.x >> 5, lane = threadIdx.x & 31;
  int ln = lane & 15, hg = lane >> 4;
  int tile = blockIdx.x * 8 + wave, base = tile * 16;
  if (base >= n_nodes) return;
  _Float16* A = Atile[wave];
  _Float16* M = Mtile[wave];

  {
    int node = base + ln;
    if (hg == 0) {
      const _Float16* gp = h16 + (size_t)node * 64;
      _Float16* lp = A + ln * 128;
#pragma unroll
      for (int c = 0; c < 64; c += 8) async_g2l_b128(lp + c, gp + c);
    } else {
      const float* p = agg + (size_t)node * 64;
#pragma unroll
      for (int c = 0; c < 64; c += 4) {
        float4 v = *(const float4*)(p + c);
        A[ln * 128 + 64 + c + 0] = (_Float16)v.x;
        A[ln * 128 + 64 + c + 1] = (_Float16)v.y;
        A[ln * 128 + 64 + c + 2] = (_Float16)v.z;
        A[ln * 128 + 64 + c + 3] = (_Float16)v.w;
      }
    }
  }
  GATHER_FENCE();

  v8f acc[4];
#pragma unroll
  for (int n = 0; n < 4; ++n) acc[n] = vzero8();
#pragma unroll
  for (int k0 = 0; k0 < 128; k0 += 32) {
    v16h af = load_a_frag(A, 128, ln, k0, hg);
#pragma unroll
    for (int n = 0; n < 4; ++n)
      acc[n] = wmma_acc(af, load_b_frag(v1t, 128, n * 16, ln, k0, hg), acc[n]);
  }
#pragma unroll
  for (int n = 0; n < 4; ++n) {
    float bias = c1[n * 16 + ln];
#pragma unroll
    for (int v = 0; v < 8; ++v) {
      float t = acc[n][v] + bias;
      M[(v + 8 * hg) * 64 + n * 16 + ln] = (_Float16)fmaxf(t, 0.f);
    }
  }
  LDS_FENCE();

  v8f acc2[4];
#pragma unroll
  for (int n = 0; n < 4; ++n) acc2[n] = vzero8();
#pragma unroll
  for (int k0 = 0; k0 < 64; k0 += 32) {
    v16h af = load_a_frag(M, 64, ln, k0, hg);
#pragma unroll
    for (int n = 0; n < 4; ++n)
      acc2[n] = wmma_acc(af, load_b_frag(v2t, 64, n * 16, ln, k0, hg), acc2[n]);
  }
#pragma unroll
  for (int n = 0; n < 4; ++n) {
    float bias = c2[n * 16 + ln];
#pragma unroll
    for (int v = 0; v < 8; ++v) {
      int row = v + 8 * hg;
      size_t idx = (size_t)(base + row) * 64 + n * 16 + ln;
      float nv = h[idx] + acc2[n][v] + bias;  // residual in f32
      h[idx] = nv;
      h16[idx] = (_Float16)nv;
    }
  }
}

// ---------------------------------------------------------------------------
// Decoder: out = relu(h @ W1 + b1) @ W2 + b2,  W2:[64,3].
// ---------------------------------------------------------------------------
__global__ void __launch_bounds__(256) decode_kernel(
    const _Float16* __restrict__ h16,
    const _Float16* __restrict__ w1t,  // [64][64]
    const float* __restrict__ b1, const float* __restrict__ w2,  // [64][3]
    const float* __restrict__ b2, float* __restrict__ out, int n_nodes) {
  __shared__ _Float16 Atile[8][16 * 64];
  __shared__ float Mid[8][16 * 64];
  int wave = threadIdx.x >> 5, lane = threadIdx.x & 31;
  int ln = lane & 15, hg = lane >> 4;
  int tile = blockIdx.x * 8 + wave, base = tile * 16;
  if (base >= n_nodes) return;
  _Float16* A = Atile[wave];
  float* M = Mid[wave];

  {
    int node = base + ln;
    const _Float16* gp = h16 + (size_t)node * 64 + hg * 32;
    _Float16* lp = A + ln * 64 + hg * 32;
#pragma unroll
    for (int c = 0; c < 32; c += 8) async_g2l_b128(lp + c, gp + c);
  }
  GATHER_FENCE();

  v8f acc[4];
#pragma unroll
  for (int n = 0; n < 4; ++n) acc[n] = vzero8();
#pragma unroll
  for (int k0 = 0; k0 < 64; k0 += 32) {
    v16h af = load_a_frag(A, 64, ln, k0, hg);
#pragma unroll
    for (int n = 0; n < 4; ++n)
      acc[n] = wmma_acc(af, load_b_frag(w1t, 64, n * 16, ln, k0, hg), acc[n]);
  }
#pragma unroll
  for (int n = 0; n < 4; ++n) {
    float bias = b1[n * 16 + ln];
#pragma unroll
    for (int v = 0; v < 8; ++v)
      M[(v + 8 * hg) * 64 + n * 16 + ln] = fmaxf(acc[n][v] + bias, 0.f);
  }
  LDS_FENCE();

  // Tiny final projection [64]->[3]: lanes0-15 do j=0,1; lanes16-31 do j=2.
  int nj = hg ? 1 : 2;
#pragma unroll
  for (int jj = 0; jj < 2; ++jj) {
    if (jj >= nj) break;
    int j = hg ? 2 : jj;
    float s = b2[j];
#pragma unroll
    for (int c = 0; c < 64; ++c) s += M[ln * 64 + c] * w2[c * 3 + j];
    out[(size_t)(base + ln) * 3 + j] = s;
  }
}

// ---------------------------------------------------------------------------
extern "C" void kernel_launch(void* const* d_in, const int* in_sizes, int n_in,
                              void* d_out, int out_size, void* d_ws,
                              size_t ws_size, hipStream_t stream) {
  const float* x      = (const float*)d_in[0];
  const int*   ei     = (const int*)d_in[1];  // [2, E]
  const float* eattr  = (const float*)d_in[2];
  const float* enc_w1 = (const float*)d_in[3];
  const float* enc_b1 = (const float*)d_in[4];
  const float* enc_w2 = (const float*)d_in[5];
  const float* enc_b2 = (const float*)d_in[6];
  const float* mw1 = (const float*)d_in[7];
  const float* mb1 = (const float*)d_in[8];
  const float* mw2 = (const float*)d_in[9];
  const float* mb2 = (const float*)d_in[10];
  const float* uw1 = (const float*)d_in[11];
  const float* ub1 = (const float*)d_in[12];
  const float* uw2 = (const float*)d_in[13];
  const float* ub2 = (const float*)d_in[14];
  const float* dec_w1 = (const float*)d_in[15];
  const float* dec_b1 = (const float*)d_in[16];
  const float* dec_w2 = (const float*)d_in[17];
  const float* dec_b2 = (const float*)d_in[18];

  const int N = in_sizes[0] / 8;   // 100000
  const int E = in_sizes[2] / 3;   // 1600000
  const int L = in_sizes[8] / 64;  // 8

  // Workspace carve-up (~65 MB): h | agg | h16 | f16 transposed weights.
  float* h   = (float*)d_ws;
  float* agg = h + (size_t)N * 64;
  _Float16* h16 = (_Float16*)(agg + (size_t)N * 64);
  _Float16* enc1t = h16 + (size_t)N * 64;
  _Float16* enc2t = enc1t + 64 * 32;
  _Float16* dec1t = enc2t + 64 * 64;
  _Float16* mw1t  = dec1t + 64 * 64;
  _Float16* mw2t  = mw1t + (size_t)L * 64 * 160;
  _Float16* uw1t  = mw2t + (size_t)L * 64 * 64;
  _Float16* uw2t  = uw1t + (size_t)L * 64 * 128;

  auto conv = [&](const float* w, _Float16* wt, int ind, int outd, int inp) {
    int tot = outd * inp;
    conv_wt_kernel<<<(tot + 255) / 256, 256, 0, stream>>>(w, wt, ind, outd,
                                                          inp);
  };
  conv(enc_w1, enc1t, 8, 64, 32);
  conv(enc_w2, enc2t, 64, 64, 64);
  conv(dec_w1, dec1t, 64, 64, 64);
  for (int l = 0; l < L; ++l) {
    conv(mw1 + (size_t)l * 131 * 64, mw1t + (size_t)l * 64 * 160, 131, 64, 160);
    conv(mw2 + (size_t)l * 64 * 64,  mw2t + (size_t)l * 64 * 64,  64, 64, 64);
    conv(uw1 + (size_t)l * 128 * 64, uw1t + (size_t)l * 64 * 128, 128, 64, 128);
    conv(uw2 + (size_t)l * 64 * 64,  uw2t + (size_t)l * 64 * 64,  64, 64, 64);
  }

  const int nodeTiles  = (N + 15) / 16;
  const int nodeBlocks = (nodeTiles + 7) / 8;
  const int edgeTiles  = (E + 15) / 16;
  const int edgeBlocks = (edgeTiles + 7) / 8;
  const int* src = ei;
  const int* dst = ei + E;
  const long long aggN = (long long)N * 64;

  encode_kernel<<<nodeBlocks, 256, 0, stream>>>(x, enc1t, enc_b1, enc2t,
                                                enc_b2, h, h16, N);
  for (int l = 0; l < L; ++l) {
    zero_f32_kernel<<<(int)((aggN + 255) / 256), 256, 0, stream>>>(agg, aggN);
    edge_msg_kernel<<<edgeBlocks, 256, 0, stream>>>(
        h16, src, dst, eattr, mw1t + (size_t)l * 64 * 160, mb1 + l * 64,
        mw2t + (size_t)l * 64 * 64, mb2 + l * 64, agg, E);
    node_upd_kernel<<<nodeBlocks, 256, 0, stream>>>(
        h, h16, agg, uw1t + (size_t)l * 64 * 128, ub1 + l * 64,
        uw2t + (size_t)l * 64 * 64, ub2 + l * 64, N);
  }
  decode_kernel<<<nodeBlocks, 256, 0, stream>>>(h16, dec1t, dec_b1, dec_w2,
                                                dec_b2, (float*)d_out, N);
}
